// MotionVQVAE_57844619542755
// MI455X (gfx1250) — compile-verified, matrix operand order checked by reading, not run on hardware
//
#include <hip/hip_runtime.h>
#include <math.h>

// ---------------------------------------------------------------------------
// CDNA5 WMMA types (wave32, v_wmma_f32_16x16x32_bf16)
// ---------------------------------------------------------------------------
typedef __attribute__((ext_vector_type(16))) __bf16 v16bf;
typedef __attribute__((ext_vector_type(8)))  float  v8f;

__device__ __forceinline__ unsigned short bf_bits(float f) {
    __bf16 h = (__bf16)f;                   // native cvt, RNE
    unsigned short u; __builtin_memcpy(&u, &h, 2); return u;
}

// Physical K index (0..31) -> position in lane-permuted fragment storage.
// Fragment element j of lane-half h maps to K = (j&7) + ((j>>3)<<4) + h*8,
// so storing K at perm(kk) makes each lane's 16 elements contiguous (32B).
__device__ __forceinline__ int perm(int kk) {
    return (kk & 7) | (((kk >> 4) & 1) << 3) | (((kk >> 3) & 1) << 4);
}

// ---------------------------------------------------------------------------
// Weight/emb converter: f32 -> bf16 in WMMA-ready permuted K-order, row-padded.
// in: (Cout, Ktot) row-major f32 ; out: (CoutPad, Ktot) bf16, K permuted /32.
// ---------------------------------------------------------------------------
__global__ void cvt_wt(const float* __restrict__ in,
                       unsigned short* __restrict__ out,
                       int Cout, int CoutPad, int Ktot)
{
    int i = blockIdx.x * 256 + threadIdx.x;
    if (i >= CoutPad * Ktot) return;
    int r = i / Ktot, k = i - r * Ktot;
    float v = (r < Cout) ? in[r * Ktot + k] : 0.f;
    out[r * Ktot + (k & ~31) + perm(k & 31)] = bf_bits(v);
}

// ---------------------------------------------------------------------------
// Implicit-GEMM conv, bf16 WMMA, compile-time shapes.
//   M = Cout, N = 16*Ho*Wo, K = Cin*KH*KW.  Block 256 thr = 8 waves (2M x 4N),
//   tile M32 x N128, 2 WMMAs / wave / K-step.
//   A fragments: direct 32B global loads from pre-permuted weights (L2).
//   B tile: im2col gather into double-buffered LDS, ONE barrier per K-step.
// act: 0=none 1=relu 2=tanh ; store_mode: 0=NCHW, 1=recon scatter into d_out
// ---------------------------------------------------------------------------
template<int Cin, int H, int W, int Cout, int KH, int KW, int STRIDE, int PAD>
__global__ __launch_bounds__(256)
void conv_wmma(const unsigned short* __restrict__ Xbf,
               const unsigned short* __restrict__ Wbf,   // permuted, row-padded
               const float* __restrict__ bias, const float* __restrict__ resid,
               float* __restrict__ Yf, unsigned short* __restrict__ Ybf,
               int act, int store_mode, int person)
{
    constexpr int Ho   = (H + 2 * PAD - KH) / STRIDE + 1;
    constexpr int Wo   = (W + 2 * PAD - KW) / STRIDE + 1;
    constexpr int KHW  = KH * KW;
    constexpr int Ktot = Cin * KHW;
    constexpr int Ntot = 16 * Ho * Wo;
    static_assert(Ktot % 32 == 0, "K tile");
    static_assert(Ntot % 128 == 0, "N tile");

    __shared__ unsigned short Bs2[2][128][48];   // [buf][n][perm(k)], 96B rows

    const int tid   = threadIdx.x;
    const int lane  = tid & 31;
    const int wid   = tid >> 5;
    const int wm    = wid >> 2;
    const int wn    = wid & 3;
    const int lhalf = lane >> 4;
    const int lrow  = lane & 15;

    const int m0 = blockIdx.y * 32;
    const int n0 = blockIdx.x * 128;

    // im2col gather of one 32-K x 128-N tile into LDS buffer `buf`
    auto stageB = [&](int buf, int kb) {
        #pragma unroll
        for (int i = 0; i < 16; ++i) {
            int e  = tid + i * 256;
            int kk = e >> 7;
            int c  = e & 127;
            int k  = kb + kk;
            int g  = n0 + c;
            int ow = g % Wo;  int t  = g / Wo;
            int oh = t % Ho;  int b  = t / Ho;
            int ci = k / KHW; int r  = k % KHW;
            int kh = r / KW;  int kw = r % KW;
            int ih = oh * STRIDE - PAD + kh;
            int iw = ow * STRIDE - PAD + kw;
            unsigned short v = 0;
            if ((PAD == 0) || (ih >= 0 && ih < H && iw >= 0 && iw < W))
                v = Xbf[((b * Cin + ci) * H + ih) * W + iw];
            Bs2[buf][c][perm(kk)] = v;
        }
    };

    // per-lane weight row pointer (row-padded alloc: always in bounds)
    const unsigned short* wrow = Wbf + (long)(m0 + wm * 16 + lrow) * Ktot;

    v8f acc0 = {0.f,0.f,0.f,0.f,0.f,0.f,0.f,0.f};
    v8f acc1 = {0.f,0.f,0.f,0.f,0.f,0.f,0.f,0.f};

    stageB(0, 0);
    __syncthreads();

    int cur = 0;
    for (int k0 = 0; k0 < Ktot; k0 += 32) {
        v16bf a, b0, b1;
        __builtin_memcpy(&a,  wrow + k0 + lhalf * 16, 32);          // global, L2
        __builtin_memcpy(&b0, &Bs2[cur][wn * 32 + lrow][lhalf * 16], 32);
        __builtin_memcpy(&b1, &Bs2[cur][wn * 32 + 16 + lrow][lhalf * 16], 32);
        if (k0 + 32 < Ktot) stageB(cur ^ 1, k0 + 32);   // prefetch next tile
        acc0 = __builtin_amdgcn_wmma_f32_16x16x32_bf16(
                   false, a, false, b0, (short)0, acc0, false, false);
        acc1 = __builtin_amdgcn_wmma_f32_16x16x32_bf16(
                   false, a, false, b1, (short)0, acc1, false, false);
        __syncthreads();
        cur ^= 1;
    }

    // ---- epilogue
    #pragma unroll
    for (int half = 0; half < 2; ++half) {
        v8f acc = half ? acc1 : acc0;
        int g  = n0 + wn * 32 + half * 16 + lrow;
        int ow = g % Wo; int t = g / Wo;
        int oh = t % Ho; int b = t / Ho;
        #pragma unroll
        for (int vr = 0; vr < 8; ++vr) {
            int row = m0 + wm * 16 + lhalf * 8 + vr;
            if (Cout % 32 == 0 || row < Cout) {
                float v = acc[vr];
                if (bias)  v += bias[row];
                long oidx = (((long)b * Cout + row) * Ho + oh) * (long)Wo + ow;
                if (resid) v += resid[oidx];
                if (act == 1)      v = v > 0.f ? v : 0.f;
                else if (act == 2) v = tanhf(v);
                if (store_mode == 0) {
                    if (Yf)  Yf[oidx]  = v;
                    if (Ybf) Ybf[oidx] = bf_bits(v);
                } else {
                    // x_recon[b, t=oh, person, ow*6+row] after leading loss slot
                    long o = 1 + (((long)b * 1024 + oh) * 2 + person) * 144
                               + (long)ow * 6 + row;
                    Yf[o] = v;
                }
            }
        }
    }
}

template<int Cin, int H, int W, int Cout, int KH, int KW, int STRIDE, int PAD>
static inline void launch_conv(const unsigned short* Xbf, const unsigned short* Wbf,
                               const float* bias, const float* resid,
                               float* Yf, unsigned short* Ybf,
                               int act, int mode, int person, hipStream_t st)
{
    constexpr int Ho = (H + 2 * PAD - KH) / STRIDE + 1;
    constexpr int Wo = (W + 2 * PAD - KW) / STRIDE + 1;
    dim3 g((16 * Ho * Wo) / 128, (Cout + 31) / 32);
    conv_wmma<Cin, H, W, Cout, KH, KW, STRIDE, PAD>
        <<<g, 256, 0, st>>>(Xbf, Wbf, bias, resid, Yf, Ybf, act, mode, person);
}

// ---------------------------------------------------------------------------
// VQ helpers
// ---------------------------------------------------------------------------
__global__ void esq_kernel(const float* __restrict__ emb, float* __restrict__ esq)
{
    int k = blockIdx.x * blockDim.x + threadIdx.x;
    if (k >= 1024) return;
    float s = 0.f;
    const float* e = emb + (long)k * 128;
    for (int d = 0; d < 128; ++d) s += e[d] * e[d];
    esq[k] = s;
}

// WMMA argmin: block = 16 z-rows x 1024 codes; dist = |z|^2 + |e|^2 - 2 z.e
// Ebf is pre-permuted: each lane's B fragment is one contiguous 32B load.
__global__ __launch_bounds__(256)
void vq_argmin(const float* __restrict__ Z, const unsigned short* __restrict__ Zbf,
               const unsigned short* __restrict__ Ebf, const float* __restrict__ esq,
               int* __restrict__ idxOut)
{
    __shared__ unsigned short zs[16][128];   // lane-permuted per 32-K group
    __shared__ float zpart[16][8];
    __shared__ float zsq[16];
    __shared__ float wdist[8][16];
    __shared__ int   widx[8][16];

    const int tid   = threadIdx.x;
    const int lane  = tid & 31;
    const int wid   = tid >> 5;
    const int lhalf = lane >> 4;
    const int lrow  = lane & 15;
    const int rowbase = blockIdx.x * 16;

    #pragma unroll
    for (int i = 0; i < 8; ++i) {
        int e = tid + i * 256;
        int r = e >> 7, d = e & 127;
        zs[r][(d & ~31) + perm(d & 31)] = Zbf[(long)(rowbase + r) * 128 + d];
    }
    if (tid < 128) {
        int r = tid >> 3, c8 = tid & 7;
        const float* z = Z + (long)(rowbase + r) * 128 + c8 * 16;
        float s = 0.f;
        #pragma unroll
        for (int d = 0; d < 16; ++d) s += z[d] * z[d];
        zpart[r][c8] = s;
    }
    __syncthreads();
    if (tid < 16) {
        float s = 0.f;
        #pragma unroll
        for (int c = 0; c < 8; ++c) s += zpart[tid][c];
        zsq[tid] = s;
    }
    __syncthreads();

    v16bf afr[4];
    #pragma unroll
    for (int s = 0; s < 4; ++s)
        __builtin_memcpy(&afr[s], &zs[lrow][s * 32 + lhalf * 16], 32);

    float best[8];  int bidx[8];
    #pragma unroll
    for (int vr = 0; vr < 8; ++vr) { best[vr] = 3.4e38f; bidx[vr] = 0; }

    for (int i = 0; i < 8; ++i) {
        int ct   = wid + i * 8;
        int code = ct * 16 + lrow;
        const unsigned short* erow = Ebf + (long)code * 128;

        v8f acc = {0.f,0.f,0.f,0.f,0.f,0.f,0.f,0.f};
        #pragma unroll
        for (int s = 0; s < 4; ++s) {
            v16bf bfr;
            __builtin_memcpy(&bfr, erow + s * 32 + lhalf * 16, 32);
            acc = __builtin_amdgcn_wmma_f32_16x16x32_bf16(
                      false, afr[s], false, bfr, (short)0, acc, false, false);
        }
        float eq = esq[code];
        #pragma unroll
        for (int vr = 0; vr < 8; ++vr) {
            int r = lhalf * 8 + vr;
            float dist = zsq[r] + eq - 2.f * acc[vr];
            if (dist < best[vr] || (dist == best[vr] && code < bidx[vr])) {
                best[vr] = dist; bidx[vr] = code;
            }
        }
    }

    #pragma unroll
    for (int off = 1; off <= 8; off <<= 1) {
        #pragma unroll
        for (int vr = 0; vr < 8; ++vr) {
            float od = __shfl_xor(best[vr], off, 32);
            int   oi = __shfl_xor(bidx[vr], off, 32);
            if (od < best[vr] || (od == best[vr] && oi < bidx[vr])) {
                best[vr] = od; bidx[vr] = oi;
            }
        }
    }
    if (lrow == 0) {
        #pragma unroll
        for (int vr = 0; vr < 8; ++vr) {
            wdist[wid][lhalf * 8 + vr] = best[vr];
            widx [wid][lhalf * 8 + vr] = bidx[vr];
        }
    }
    __syncthreads();
    if (tid < 16) {
        float bd = wdist[0][tid]; int bi = widx[0][tid];
        for (int w = 1; w < 8; ++w) {
            float d = wdist[w][tid]; int i2 = widx[w][tid];
            if (d < bd || (d == bd && i2 < bi)) { bd = d; bi = i2; }
        }
        idxOut[rowbase + tid] = bi;
    }
}

// q = emb[idx] (bf16 shadow for decoder), sum((q-z)^2), histogram.
__global__ __launch_bounds__(128)
void vq_apply(const float* __restrict__ Z, const float* __restrict__ emb,
              const int* __restrict__ idx, unsigned short* __restrict__ Qbf,
              float* __restrict__ lossAcc, float* __restrict__ hist)
{
    __shared__ float red[128];
    const int n = blockIdx.x;
    const int t = threadIdx.x;
    const int k = idx[n];
    float q = emb[(long)k * 128 + t];
    Qbf[(long)n * 128 + t] = bf_bits(q);
    float d = q - Z[(long)n * 128 + t];
    red[t] = d * d;
    __syncthreads();
    for (int s = 64; s > 0; s >>= 1) {
        if (t < s) red[t] += red[t + s];
        __syncthreads();
    }
    if (t == 0) {
        atomicAdd(lossAcc, red[0]);
        atomicAdd(hist + k, 1.0f);
    }
}

// ---------------------------------------------------------------------------
// 2x bilinear upsample (reference align-corners weights), f32 in -> bf16 out
// ---------------------------------------------------------------------------
__global__ void upsample2(const float* __restrict__ in,
                          unsigned short* __restrict__ outbf,
                          int BC, int H, int W)
{
    long total = (long)BC * 4 * H * W;
    long i = (long)blockIdx.x * blockDim.x + threadIdx.x;
    if (i >= total) return;
    int W2 = 2 * W, H2 = 2 * H;
    int ow = (int)(i % W2); long t = i / W2;
    int oh = (int)(t % H2); int bc = (int)(t / H2);
    float sH = (H - 1.0f) / (float)(H2 - 1);
    float sW = (W - 1.0f) / (float)(W2 - 1);
    float ph = oh * sH, pw = ow * sW;
    int hl = (int)floorf(ph); int hh = hl + 1 < H ? hl + 1 : H - 1; float fh = ph - hl;
    int wl = (int)floorf(pw); int wh = wl + 1 < W ? wl + 1 : W - 1; float fw = pw - wl;
    const float* base = in + (long)bc * H * W;
    float v00 = base[(long)hl * W + wl], v01 = base[(long)hl * W + wh];
    float v10 = base[(long)hh * W + wl], v11 = base[(long)hh * W + wh];
    float v = (1.f - fh) * ((1.f - fw) * v00 + fw * v01)
            +         fh * ((1.f - fw) * v10 + fw * v11);
    outbf[i] = bf_bits(v);
}

// person slice (B,T,2,144) -> NCHW (B,6,1024,24), bf16 shadow only
__global__ void extract_person(const float* __restrict__ x,
                               unsigned short* __restrict__ pbf, int p)
{
    long i = (long)blockIdx.x * blockDim.x + threadIdx.x;
    if (i >= 2359296L) return;
    int j = (int)(i % 24); long t1 = i / 24;
    int t = (int)(t1 % 1024); long t2 = t1 / 1024;
    int d = (int)(t2 % 6); int b = (int)(t2 / 6);
    pbf[i] = bf_bits(x[(((long)b * 1024 + t) * 2 + p) * 144 + (long)j * 6 + d]);
}

// loss + perplexities
__global__ void finalize_kernel(const float* __restrict__ lossAcc,
                                const float* __restrict__ hist,
                                float* __restrict__ out, int Nrows, long Relems)
{
    __shared__ float red[1024];
    int t = threadIdx.x;
    for (int p = 0; p < 2; ++p) {
        float c  = hist[p * 1024 + t];
        float pr = c / (float)Nrows;
        red[t] = -pr * logf(pr + 1e-10f);
        __syncthreads();
        for (int s = 512; s > 0; s >>= 1) {
            if (t < s) red[t] += red[t + s];
            __syncthreads();
        }
        if (t == 0) out[1 + Relems + p] = expf(red[0]);
        __syncthreads();
    }
    if (t == 0) out[0] = 0.55f * (lossAcc[0] + lossAcc[1]) / ((float)Nrows * 128.f);
}

// ---------------------------------------------------------------------------
// Host-side orchestration
// ---------------------------------------------------------------------------
static inline void cvtw(const float* in, unsigned short* out,
                        int Cout, int CoutPad, int Ktot, hipStream_t st)
{
    int n = CoutPad * Ktot;
    cvt_wt<<<(n + 255) / 256, 256, 0, st>>>(in, out, Cout, CoutPad, Ktot);
}

extern "C" void kernel_launch(void* const* d_in, const int* in_sizes, int n_in,
                              void* d_out, int out_size, void* d_ws, size_t ws_size,
                              hipStream_t stream)
{
    const float* x    = (const float*)d_in[0];
    const float* e_w1 = (const float*)d_in[1];
    const float* e_b1 = (const float*)d_in[2];
    const float* e_w2 = (const float*)d_in[3];
    const float* e_b2 = (const float*)d_in[4];
    const float* e_w3 = (const float*)d_in[5];
    const float* e_b3 = (const float*)d_in[6];
    const float* er_c1[3] = {(const float*)d_in[7],  (const float*)d_in[9],  (const float*)d_in[11]};
    const float* er_c2[3] = {(const float*)d_in[8],  (const float*)d_in[10], (const float*)d_in[12]};
    const float* pre_w = (const float*)d_in[13];
    const float* pre_b = (const float*)d_in[14];
    const float* emb   = (const float*)d_in[15];
    const float* d_w1  = (const float*)d_in[16];
    const float* d_b1  = (const float*)d_in[17];
    const float* dr_c1[3] = {(const float*)d_in[18], (const float*)d_in[20], (const float*)d_in[22]};
    const float* dr_c2[3] = {(const float*)d_in[19], (const float*)d_in[21], (const float*)d_in[23]};
    const float* d_wu1 = (const float*)d_in[24];
    const float* d_bu1 = (const float*)d_in[25];
    const float* d_wu2 = (const float*)d_in[26];
    const float* d_bu2 = (const float*)d_in[27];

    float* ws  = (float*)d_ws;
    float* out = (float*)d_out;

    // ---- f32 region (floats)
    const size_t DF   = 0;               // 6291456   conv3/res ping (also decoder)
    const size_t EF   = DF  + 6291456;   // 6291456   res pong
    const size_t ZF   = EF  + 6291456;   // 3145728   pre-vq latents
    const size_t W1F  = ZF  + 3145728;   // 12582912  d_wu1 out (f32 for upsample)
    const size_t ESQ  = W1F + 12582912;  // 1024
    const size_t ACC  = ESQ + 1024;      // 2
    const size_t HIST = ACC + 2;         // 2048
    const size_t FTOT = HIST + 2048;

    float* Dp   = ws + DF;
    float* Ep   = ws + EF;
    float* Zb   = ws + ZF;
    float* W1   = ws + W1F;
    float* esqB = ws + ESQ;
    float* acc  = ws + ACC;
    float* hist = ws + HIST;

    // ---- bf16 region (ushorts)
    unsigned short* bs = (unsigned short*)(ws + FTOT);
    const size_t PBF  = 0;                // 2359296  person
    const size_t C1BF = PBF  + 2359296;   // 12582912
    const size_t C2BF = C1BF + 12582912;  // 6291456
    const size_t DBF  = C2BF + 6291456;   // 6291456
    const size_t EBF  = DBF  + 6291456;   // 6291456
    const size_t HBF  = EBF  + 6291456;   // 1572864
    const size_t ZBF  = HBF  + 1572864;   // 3145728
    const size_t QBF  = ZBF  + 3145728;   // 3145728
    const size_t U1BF = QBF  + 3145728;   // 25165824
    const size_t U2BF = U1BF + 25165824;  // 50331648
    const size_t WBF  = U2BF + 50331648;  // weights(padded)+emb: 2899968
    const size_t BTOT = WBF  + 2899968;

    unsigned short* pbf  = bs + PBF;
    unsigned short* c1bf = bs + C1BF;
    unsigned short* c2bf = bs + C2BF;
    unsigned short* dbf  = bs + DBF;
    unsigned short* ebf  = bs + EBF;
    unsigned short* hbf  = bs + HBF;
    unsigned short* zbf  = bs + ZBF;
    unsigned short* qbf  = bs + QBF;
    unsigned short* u1bf = bs + U1BF;
    unsigned short* u2bf = bs + U2BF;

    // weight bf16 sub-offsets (permuted layouts; d_wu2 padded 6 -> 32 rows)
    unsigned short* cur = bs + WBF;
    unsigned short* w_e1  = cur; cur += 12288;    // 128 x 96
    unsigned short* w_e2  = cur; cur += 524288;   // 256 x 2048
    unsigned short* w_e3  = cur; cur += 589824;   // 256 x 2304
    unsigned short* w_ec1[3]; unsigned short* w_ec2[3];
    for (int l = 0; l < 3; ++l) { w_ec1[l] = cur; cur += 147456;   // 64 x 2304
                                  w_ec2[l] = cur; cur += 16384; }  // 256 x 64
    unsigned short* w_pre = cur; cur += 32768;    // 128 x 256
    unsigned short* w_d1  = cur; cur += 294912;   // 256 x 1152
    unsigned short* w_dc1[3]; unsigned short* w_dc2[3];
    for (int l = 0; l < 3; ++l) { w_dc1[l] = cur; cur += 147456;
                                  w_dc2[l] = cur; cur += 16384; }
    unsigned short* w_du1 = cur; cur += 294912;   // 128 x 2304
    unsigned short* w_du2 = cur; cur += 36864;    // 32(pad) x 1152
    unsigned short* embbf = cur; cur += 131072;   // 1024 x 128

    int* idxB = (int*)(bs + BTOT + (BTOT & 1));

    const int  NROWS = 24576;
    const long RECON = 4718592L;

    // ---- one-time per launch: zero accumulators, pre-convert weights + emb
    hipMemsetAsync(acc, 0, (2 + 2048) * sizeof(float), stream);
    cvtw(e_w1, w_e1, 128, 128,   96, stream);
    cvtw(e_w2, w_e2, 256, 256, 2048, stream);
    cvtw(e_w3, w_e3, 256, 256, 2304, stream);
    for (int l = 0; l < 3; ++l) { cvtw(er_c1[l], w_ec1[l],  64,  64, 2304, stream);
                                  cvtw(er_c2[l], w_ec2[l], 256, 256,   64, stream); }
    cvtw(pre_w, w_pre, 128, 128,  256, stream);
    cvtw(d_w1,  w_d1,  256, 256, 1152, stream);
    for (int l = 0; l < 3; ++l) { cvtw(dr_c1[l], w_dc1[l],  64,  64, 2304, stream);
                                  cvtw(dr_c2[l], w_dc2[l], 256, 256,   64, stream); }
    cvtw(d_wu1, w_du1, 128, 128, 2304, stream);
    cvtw(d_wu2, w_du2,   6,  32, 1152, stream);
    cvtw(emb,   embbf, 1024, 1024, 128, stream);
    esq_kernel<<<4, 256, 0, stream>>>(emb, esqB);

    for (int p = 0; p < 2; ++p) {
        extract_person<<<(2359296 + 255) / 256, 256, 0, stream>>>(x, pbf, p);

        // ---------------- encoder ----------------
        launch_conv<  6, 1024, 24, 128, 4, 4, 2, 1>(pbf,  w_e1, e_b1, nullptr, nullptr, c1bf, 1, 0, 0, stream);
        launch_conv<128,  512, 12, 256, 4, 4, 2, 1>(c1bf, w_e2, e_b2, nullptr, nullptr, c2bf, 1, 0, 0, stream);
        launch_conv<256,  256,  6, 256, 3, 3, 1, 1>(c2bf, w_e3, e_b3, nullptr, Dp,      dbf,  0, 0, 0, stream);
        {
            float* srcF = Dp; unsigned short* srcB = dbf;
            float* dstF = Ep; unsigned short* dstB = ebf;
            for (int l = 0; l < 3; ++l) {
                launch_conv<256, 256, 6,  64, 3, 3, 1, 1>(srcB, w_ec1[l], nullptr, nullptr, nullptr, hbf, 1, 0, 0, stream);
                launch_conv< 64, 256, 6, 256, 1, 1, 1, 0>(hbf,  w_ec2[l], nullptr, srcF,    dstF,    dstB,1, 0, 0, stream);
                float* tf = srcF; srcF = dstF; dstF = tf;
                unsigned short* tb = srcB; srcB = dstB; dstB = tb;
            }
            // result lives in (Ep, ebf)
        }
        launch_conv<256, 256, 6, 128, 1, 1, 1, 0>(ebf, w_pre, pre_b, nullptr, Zb, zbf, 0, 0, 0, stream);

        // ---------------- VQ ----------------
        vq_argmin<<<NROWS / 16, 256, 0, stream>>>(Zb, zbf, embbf, esqB, idxB);
        vq_apply<<<NROWS, 128, 0, stream>>>(Zb, emb, idxB, qbf, acc + p, hist + p * 1024);

        // ---------------- decoder ----------------
        launch_conv<128, 256, 6, 256, 3, 3, 1, 1>(qbf, w_d1, d_b1, nullptr, Dp, dbf, 0, 0, 0, stream);
        {
            float* srcF = Dp; unsigned short* srcB = dbf;
            float* dstF = Ep; unsigned short* dstB = ebf;
            for (int l = 0; l < 3; ++l) {
                launch_conv<256, 256, 6,  64, 3, 3, 1, 1>(srcB, w_dc1[l], nullptr, nullptr, nullptr, hbf, 1, 0, 0, stream);
                launch_conv< 64, 256, 6, 256, 1, 1, 1, 0>(hbf,  w_dc2[l], nullptr, srcF,    dstF,    dstB,1, 0, 0, stream);
                float* tf = srcF; srcF = dstF; dstF = tf;
                unsigned short* tb = srcB; srcB = dstB; dstB = tb;
            }
        }
        upsample2<<<(25165824 + 255) / 256, 256, 0, stream>>>(Ep, u1bf, 16 * 256, 256, 6);
        launch_conv<256,  512, 12, 128, 3, 3, 1, 1>(u1bf, w_du1, d_bu1, nullptr, W1, nullptr, 0, 0, 0, stream);
        upsample2<<<(50331648 + 255) / 256, 256, 0, stream>>>(W1, u2bf, 16 * 128, 512, 12);
        launch_conv<128, 1024, 24,   6, 3, 3, 1, 1>(u2bf, w_du2, d_bu2, nullptr, out, nullptr, 2, 1, p, stream);
    }

    finalize_kernel<<<1, 1024, 0, stream>>>(acc, hist, out, NROWS, RECON);
}
